// _ConvEmformerLayer_89945205113519
// MI455X (gfx1250) — matrix operand order repaired and checked by reading, not compiled, not run on hardware
//
#include <hip/hip_runtime.h>

// ---------------------------------------------------------------------------
// ConvEmformer layer forward for MI455X (gfx1250, wave32, WMMA).
// - All dense projections on v_wmma_f32_16x16x32_bf16 (bf16 operands, f32 acc)
// - bf16 tiles staged with GLOBAL_LOAD_ASYNC_TO_LDS_B128 (ASYNCcnt), software
//   pipelined with double-buffered LDS: s_wait_asynccnt 4 retires only the
//   older tile while the next tile's loads stay in flight.
// - LDS tiles in *fragment order*: each lane's 16-element WMMA fragment is 32
//   contiguous bytes -> 2x ds_load_b128, zero repack VALU ops.
// - Weights converted to bf16 once per launch; activation bf16 shadows fused
//   into LN / SiLU / attention / GEMM epilogues. A-scratch padded to 128-row
//   multiples so staging is unconditional (no exec-mask juggling).
// ---------------------------------------------------------------------------

#define D_ 1024
#define F_ 4096
#define H_ 16
#define DH_ 64
#define B_ 4
#define T_ 1024
#define R_ 256
#define S_ 8
#define M_ 4
#define NS_ 8
#define SEG_ 128
#define RC_ 32
#define KSZ_ 31
#define NTOK 1280          /* R + T */
#define QTOK 1288          /* R + T + S */
#define KLTOK 1284         /* M + R + T */
#define KL_MAX 1312
#define PADROWS 5248       /* QTOK*B_ (5152) padded to a multiple of 128 */

typedef __attribute__((ext_vector_type(16))) __bf16 v16bf;
typedef __attribute__((ext_vector_type(8)))  float  v8f;
typedef int v4i_g __attribute__((vector_size(16)));   // matches builtin param type

#if defined(__has_builtin)
#if __has_builtin(__builtin_amdgcn_global_load_async_to_lds_b128)
#define USE_ASYNC_LDS 1
#endif
#endif

__device__ __forceinline__ unsigned short f2bf_bits(float f) {
  union { float f; unsigned u; } v; v.f = f;
  unsigned r = v.u + 0x7FFFu + ((v.u >> 16) & 1u);   // round-to-nearest-even
  return (unsigned short)(r >> 16);
}

// Copy 16 bytes global -> LDS (async on gfx1250).
__device__ __forceinline__ void stage16(const unsigned short* g, unsigned short* l) {
#ifdef USE_ASYNC_LDS
  __builtin_amdgcn_global_load_async_to_lds_b128(
      (__attribute__((address_space(1))) v4i_g*)(size_t)g,
      (__attribute__((address_space(3))) v4i_g*)(unsigned int)(size_t)l,
      0, 0);
#else
  *reinterpret_cast<uint4*>(l) = *reinterpret_cast<const uint4*>(g);
#endif
}

// ---------------------------------------------------------------------------
// GEMM: C[M,N] = act(Abf[M,K](bf16) * Wbf[N,K]^T(bf16) + bias) * alpha
//                + (row < res_rows ? res : 0)
// Optional bf16 shadow output Cbf; fp32 output C may be null.
// BM=BN=128, BK=32. 256 threads = 8 waves; wave = 2x4 grid of 16x16 WMMA.
// Abf must be readable up to ceil(M/128)*128 rows (padded scratch).
// LDS fragment-order layout: ls[subtile(0..7)][lane(0..31)][j(0..15)].
//   A: (row,k) -> lane=(row&15)|(((k>>3)&1)<<4), j=((k>>4)<<3)|(k&7)
//   B: (n,  k) -> lane=(n&15)|(((k>>3)&1)<<4), j=((k>>4)<<3)|(k&7)
// Both keep 8-wide K-chunks contiguous in j => 16B global load == 16B LDS store.
// ---------------------------------------------------------------------------
__global__ __launch_bounds__(256)
void gemm_bf16_wmma(const unsigned short* __restrict__ Abf,
                    const unsigned short* __restrict__ Wbf,
                    const float* __restrict__ bias,
                    const float* __restrict__ res, int res_rows,
                    float* __restrict__ C,
                    unsigned short* __restrict__ Cbf,
                    int M, int N, int K, float alpha, int relu) {
  __shared__ unsigned short lsA[2][4096];   // ping-pong: 8 subtiles*32 lanes*16
  __shared__ unsigned short lsB[2][4096];

  const int tid  = threadIdx.x;
  const int lane = tid & 31;
  const int wave = tid >> 5;
  const int m0 = blockIdx.y * 128;
  const int n0 = blockIdx.x * 128;
  const int amb = (wave >> 1) * 2;   // wave's A subtile base (row/16)
  const int bnb = (wave & 1) * 4;    // wave's B subtile base (col/16)
  const int hl  = lane >> 4;
  const int l15 = lane & 15;

  // Per-thread staging geometry: chunks c = tid and tid+256 of 512 per matrix.
  const int rowc0 = tid >> 2,          rowc1 = (tid + 256) >> 2;
  const int k0c0  = (tid & 3) * 8,     k0c1  = ((tid + 256) & 3) * 8;
  const int dA0 = (((rowc0 >> 4) * 32 + ((rowc0 & 15) | (((k0c0 >> 3) & 1) << 4))) << 4)
                  + ((k0c0 >> 4) << 3);
  const int dA1 = (((rowc1 >> 4) * 32 + ((rowc1 & 15) | (((k0c1 >> 3) & 1) << 4))) << 4)
                  + ((k0c1 >> 4) << 3);
  const int dB0 = (((rowc0 >> 4) * 32 + ((rowc0 & 15) | (((k0c0 >> 3) & 1) << 4))) << 4)
                  + ((k0c0 >> 4) << 3);
  const int dB1 = (((rowc1 >> 4) * 32 + ((rowc1 & 15) | (((k0c1 >> 3) & 1) << 4))) << 4)
                  + ((k0c1 >> 4) << 3);
  const unsigned short* gA0 = &Abf[(size_t)(m0 + rowc0) * K + k0c0];
  const unsigned short* gA1 = &Abf[(size_t)(m0 + rowc1) * K + k0c1];
  const unsigned short* gB0 = &Wbf[(size_t)(n0 + rowc0) * K + k0c0];
  const unsigned short* gB1 = &Wbf[(size_t)(n0 + rowc1) * K + k0c1];

  v8f acc[2][4];
#pragma unroll
  for (int r = 0; r < 2; ++r)
#pragma unroll
    for (int c = 0; c < 4; ++c) acc[r][c] = (v8f){0.f,0.f,0.f,0.f,0.f,0.f,0.f,0.f};

  // prologue: stage tile 0 into buffer 0 (4 async instructions per wave)
  stage16(gA0, &lsA[0][dA0]);
  stage16(gA1, &lsA[0][dA1]);
  stage16(gB0, &lsB[0][dB0]);
  stage16(gB1, &lsB[0][dB1]);

  int buf = 0;
  const int niter = K >> 5;
  for (int it = 0; it < niter; ++it) {
    const int ktn = (it + 1) << 5;
    const bool more = ktn < K;
    __syncthreads();                 // all waves done reading buf^1 (prev iter)
    if (more) {                      // stage next tile into the other buffer
      stage16(gA0 + ktn, &lsA[buf ^ 1][dA0]);
      stage16(gA1 + ktn, &lsA[buf ^ 1][dA1]);
      stage16(gB0 + ktn, &lsB[buf ^ 1][dB0]);
      stage16(gB1 + ktn, &lsB[buf ^ 1][dB1]);
#ifdef USE_ASYNC_LDS
      __builtin_amdgcn_s_wait_asynccnt(4);   // retire older tile only
#endif
    } else {
#ifdef USE_ASYNC_LDS
      __builtin_amdgcn_s_wait_asynccnt(0);
#endif
    }
    __syncthreads();                 // everyone's buf data visible

    v16bf aF[2], bF[4];
#pragma unroll
    for (int r = 0; r < 2; ++r)
      aF[r] = *reinterpret_cast<const v16bf*>(&lsA[buf][((amb + r) * 32 + lane) << 4]);
#pragma unroll
    for (int c = 0; c < 4; ++c)
      bF[c] = *reinterpret_cast<const v16bf*>(&lsB[buf][((bnb + c) * 32 + lane) << 4]);
#pragma unroll
    for (int r = 0; r < 2; ++r)
#pragma unroll
      for (int c = 0; c < 4; ++c)
        acc[r][c] = __builtin_amdgcn_wmma_f32_16x16x32_bf16(
            false, aF[r], false, bF[c], (short)0, acc[r][c], false, false);
    buf ^= 1;
  }

  // Epilogue: C/D 16x16 f32 layout -> row = base + hl*8 + v, col = base + l15
#pragma unroll
  for (int r = 0; r < 2; ++r)
#pragma unroll
    for (int c = 0; c < 4; ++c)
#pragma unroll
      for (int v = 0; v < 8; ++v) {
        const int row = m0 + amb * 16 + r * 16 + hl * 8 + v;
        const int col = n0 + bnb * 16 + c * 16 + l15;
        if (row < M) {
          float val = acc[r][c][v];
          if (bias) val += bias[col];
          if (relu) val = fmaxf(val, 0.f);
          val *= alpha;
          if (res && row < res_rows) val += res[(size_t)row * N + col];
          if (C)   C[(size_t)row * N + col] = val;
          if (Cbf) Cbf[(size_t)row * N + col] = f2bf_bits(val);
        }
      }
}

// fp32 -> bf16 bits conversion
__global__ void tobf16_kernel(const float* __restrict__ x,
                              unsigned short* __restrict__ y, size_t n) {
  const size_t i = (size_t)blockIdx.x * 256 + threadIdx.x;
  if (i < n) y[i] = f2bf_bits(x[i]);
}

// ---------------------------------------------------------------------------
// Row LayerNorm over D=1024, one block per row; optional bf16 shadow output.
// ---------------------------------------------------------------------------
__global__ __launch_bounds__(256)
void ln_kernel(const float* __restrict__ x, const float* __restrict__ g,
               const float* __restrict__ b, float* __restrict__ y,
               unsigned short* __restrict__ ybf, int nrows) {
  const int row = blockIdx.x;
  if (row >= nrows) return;
  const float* xr = x + (size_t)row * D_;
  __shared__ float red[256];
  float v[4]; float s = 0.f;
#pragma unroll
  for (int i = 0; i < 4; ++i) { v[i] = xr[threadIdx.x + i * 256]; s += v[i]; }
  red[threadIdx.x] = s; __syncthreads();
  for (int off = 128; off > 0; off >>= 1) {
    if ((int)threadIdx.x < off) red[threadIdx.x] += red[threadIdx.x + off];
    __syncthreads();
  }
  const float mu = red[0] * (1.f / D_);
  __syncthreads();
  float s2 = 0.f;
#pragma unroll
  for (int i = 0; i < 4; ++i) { float d = v[i] - mu; s2 += d * d; }
  red[threadIdx.x] = s2; __syncthreads();
  for (int off = 128; off > 0; off >>= 1) {
    if ((int)threadIdx.x < off) red[threadIdx.x] += red[threadIdx.x + off];
    __syncthreads();
  }
  const float inv = rsqrtf(red[0] * (1.f / D_) + 1e-5f);
#pragma unroll
  for (int i = 0; i < 4; ++i) {
    const int c = threadIdx.x + i * 256;
    const float val = (v[i] - mu) * inv * g[c] + b[c];
    y[(size_t)row * D_ + c] = val;
    if (ybf) ybf[(size_t)row * D_ + c] = f2bf_bits(val);
  }
}

// summary[s,b,d] = mean_t ln_x[R + s*SEG + t, b, d], appended at qin row NTOK+s
__global__ void summary_kernel(const float* __restrict__ lnx, float* __restrict__ qin) {
  const int i = blockIdx.x * 256 + threadIdx.x;
  if (i >= NS_ * B_ * D_) return;
  const int d = i & (D_ - 1);
  const int b = (i >> 10) & (B_ - 1);
  const int s = i >> 12;
  float acc = 0.f;
  for (int t = 0; t < SEG_; ++t)
    acc += lnx[(((size_t)(R_ + s * SEG_ + t)) * B_ + b) * D_ + d];
  qin[(((size_t)(NTOK + s)) * B_ + b) * D_ + d] = acc * (1.f / SEG_);
}

// ---------------------------------------------------------------------------
// Fused attention: block = (q token, batch); 8 waves x 2 heads.
// Writes fp32 output + bf16 shadow (input to the out-projection GEMM).
// ---------------------------------------------------------------------------
__global__ __launch_bounds__(256)
void attention_kernel(const float* __restrict__ q, const float* __restrict__ kv,
                      const unsigned char* __restrict__ mask,
                      const int* __restrict__ lengths,
                      float* __restrict__ out, unsigned short* __restrict__ outbf,
                      int Q, int KL) {
  const int qi = blockIdx.x;
  const int b  = blockIdx.y;
  const int tid = threadIdx.x, lane = tid & 31, wave = tid >> 5;
  __shared__ float sQ[D_];
  __shared__ float sP[8][KL_MAX];
  const float* qr = q + ((size_t)qi * B_ + b) * D_;
  for (int i = tid; i < D_; i += 256) sQ[i] = qr[i] * 0.125f;  // DH^-0.5
  __syncthreads();
  int maxlen = 0;
  for (int i = 0; i < B_; ++i) maxlen = max(maxlen, lengths[i]);
  const int klen = lengths[b] + M_ + (Q - maxlen - S_);
  for (int hh = 0; hh < 2; ++hh) {
    const int h = wave * 2 + hh;
    const float* qh = sQ + h * DH_;
    float lmax = -3.0e38f;
    for (int k = lane; k < KL; k += 32) {
      const float* kr = kv + ((size_t)k * B_ + b) * (2 * D_) + h * DH_;
      float s = 0.f;
#pragma unroll 8
      for (int d = 0; d < DH_; ++d) s += qh[d] * kr[d];
      if (mask[(size_t)qi * KL + k]) s = -1.0e8f;
      if (k >= klen) s = -1.0e8f;
      sP[wave][k] = s;
      lmax = fmaxf(lmax, s);
    }
#pragma unroll
    for (int o = 16; o > 0; o >>= 1) lmax = fmaxf(lmax, __shfl_xor(lmax, o, 32));
    float lsum = 0.f;
    for (int k = lane; k < KL; k += 32) {
      const float p = __expf(sP[wave][k] - lmax);
      sP[wave][k] = p;
      lsum += p;
    }
#pragma unroll
    for (int o = 16; o > 0; o >>= 1) lsum += __shfl_xor(lsum, o, 32);
    const float inv = 1.f / lsum;
    float a0 = 0.f, a1 = 0.f;
    for (int k = 0; k < KL; ++k) {
      const float p = sP[wave][k];
      const float* vr = kv + ((size_t)k * B_ + b) * (2 * D_) + D_ + h * DH_;
      a0 += p * vr[lane];
      a1 += p * vr[lane + 32];
    }
    const size_t o0 = ((size_t)qi * B_ + b) * D_ + h * DH_;
    out[o0 + lane]        = a0 * inv;
    out[o0 + lane + 32]   = a1 * inv;
    outbf[o0 + lane]      = f2bf_bits(a0 * inv);
    outbf[o0 + lane + 32] = f2bf_bits(a1 * inv);
  }
}

__global__ void clip_mems_kernel(const float* __restrict__ aproj, float* __restrict__ dst) {
  const int i = blockIdx.x * 256 + threadIdx.x;
  if (i >= S_ * B_ * D_) return;
  const float v = aproj[(size_t)(QTOK - S_) * B_ * D_ + i];
  dst[i] = fminf(fmaxf(v, -10.f), 10.f);
}

__global__ void glu_kernel(const float* __restrict__ x, float* __restrict__ y, int rows) {
  const size_t i = (size_t)blockIdx.x * 256 + threadIdx.x;
  if (i >= (size_t)rows * D_) return;
  const size_t r = i >> 10; const int d = (int)(i & (D_ - 1));
  const float a = x[r * (2 * D_) + d];
  const float g = x[r * (2 * D_) + D_ + d];
  y[i] = a * (1.f / (1.f + __expf(-g)));
}

// in-place SiLU + bf16 shadow
__global__ void silu_kernel(float* __restrict__ x, unsigned short* __restrict__ xbf,
                            size_t n) {
  const size_t i = (size_t)blockIdx.x * 256 + threadIdx.x;
  if (i < n) {
    const float v = x[i];
    const float y = v * (1.f / (1.f + __expf(-v)));
    x[i] = y;
    xbf[i] = f2bf_bits(y);
  }
}

__global__ void conv_utt_kernel(const float* __restrict__ x, const float* __restrict__ w,
                                const float* __restrict__ bias, float* __restrict__ y) {
  const size_t i = (size_t)blockIdx.x * 256 + threadIdx.x;
  if (i >= (size_t)T_ * B_ * D_) return;
  const int d = (int)(i & (D_ - 1));
  const int b = (int)((i >> 10) & (B_ - 1));
  const int t = (int)(i >> 12);
  float acc = bias[d];
#pragma unroll
  for (int k = 0; k < KSZ_; ++k) {
    const int tt = t - (KSZ_ - 1) + k;
    if (tt >= 0)
      acc += w[d * KSZ_ + k] * x[(((size_t)(R_ + tt)) * B_ + b) * D_ + d];
  }
  y[(((size_t)(R_ + t)) * B_ + b) * D_ + d] = acc;
}

__global__ void conv_rc_kernel(const float* __restrict__ x, const float* __restrict__ w,
                               const float* __restrict__ bias, float* __restrict__ y) {
  const size_t i = (size_t)blockIdx.x * 256 + threadIdx.x;
  if (i >= (size_t)R_ * B_ * D_) return;
  const int d  = (int)(i & (D_ - 1));
  const int b  = (int)((i >> 10) & (B_ - 1));
  const int rr = (int)(i >> 12);
  const int s = rr >> 5;
  const int r = rr & 31;
  float acc = bias[d];
#pragma unroll
  for (int k = 0; k < KSZ_; ++k) {
    const int j = r + k;
    float xv;
    if (j < KSZ_ - 1)
      xv = x[(((size_t)(R_ + (s + 1) * SEG_ - (KSZ_ - 1) + j)) * B_ + b) * D_ + d];
    else
      xv = x[(((size_t)(s * RC_ + (j - (KSZ_ - 1)))) * B_ + b) * D_ + d];
    acc += w[d * KSZ_ + k] * xv;
  }
  y[(((size_t)(s * RC_ + r)) * B_ + b) * D_ + d] = acc;
}

// ---------------------------------------------------------------------------
// Host-side orchestration
// ---------------------------------------------------------------------------
static inline dim3 gemm_grid(int M, int N) { return dim3((N + 127) / 128, (M + 127) / 128); }
static inline unsigned blks(size_t n) { return (unsigned)((n + 255) / 256); }

extern "C" void kernel_launch(void* const* d_in, const int* in_sizes, int n_in,
                              void* d_out, int out_size, void* d_ws, size_t ws_size,
                              hipStream_t stream) {
  (void)in_sizes; (void)n_in; (void)out_size; (void)ws_size;
  const float* utter   = (const float*)d_in[0];
  const int*   lengths = (const int*)d_in[1];
  const float* rc      = (const float*)d_in[2];
  const float* mems    = (const float*)d_in[3];
  const unsigned char* amask = (const unsigned char*)d_in[4];
  const float* f0_ln_g = (const float*)d_in[5];
  const float* f0_ln_b = (const float*)d_in[6];
  const float* f0_w_in = (const float*)d_in[7];
  const float* f0_b_in = (const float*)d_in[8];
  const float* f0_w_out= (const float*)d_in[9];
  const float* f0_b_out= (const float*)d_in[10];
  const float* f1_ln_g = (const float*)d_in[11];
  const float* f1_ln_b = (const float*)d_in[12];
  const float* f1_w_in = (const float*)d_in[13];
  const float* f1_b_in = (const float*)d_in[14];
  const float* f1_w_out= (const float*)d_in[15];
  const float* f1_b_out= (const float*)d_in[16];
  const float* ln_in_g = (const float*)d_in[17];
  const float* ln_in_b = (const float*)d_in[18];
  const float* ln_out_g= (const float*)d_in[19];
  const float* ln_out_b= (const float*)d_in[20];
  const float* wq      = (const float*)d_in[21];
  const float* bq      = (const float*)d_in[22];
  const float* wkv     = (const float*)d_in[23];
  const float* bkv     = (const float*)d_in[24];
  const float* wout    = (const float*)d_in[25];
  const float* bout    = (const float*)d_in[26];
  const float* c_pre_ln_g = (const float*)d_in[27];
  const float* c_pre_ln_b = (const float*)d_in[28];
  const float* c_pre_w    = (const float*)d_in[29];
  const float* c_pre_b    = (const float*)d_in[30];
  const float* c_conv_w   = (const float*)d_in[31];
  const float* c_conv_b   = (const float*)d_in[32];
  const float* c_post_ln_g= (const float*)d_in[33];
  const float* c_post_ln_b= (const float*)d_in[34];
  const float* c_post_w   = (const float*)d_in[35];
  const float* c_post_b   = (const float*)d_in[36];

  float* ws  = (float*)d_ws;
  float* out = (float*)d_out;

  const size_t SZ_TOK  = (size_t)NTOK  * B_ * D_;       // 5,242,880
  const size_t SZ_Q    = (size_t)QTOK  * B_ * D_;       // 5,275,648
  const size_t SZ_KVIN = (size_t)KLTOK * B_ * D_;       // 5,259,264
  const size_t SZ_KVO  = (size_t)KLTOK * B_ * (2 * D_); // 10,518,528
  const size_t SZ_BIG  = (size_t)NTOK  * B_ * F_;       // 20,971,520
  const size_t SZ_ABF  = (size_t)PADROWS * D_;          // 5,373,952 (padded)

  // fp32 working buffers (aliased over their lifetimes)
  float* b_inp   = ws;
  float* b_ffn0  = b_inp  + SZ_TOK;
  float* b_lnx   = b_ffn0 + SZ_TOK;      // later: res3
  float* b_big   = b_lnx  + SZ_TOK;      // conv pre-projection fp32 output
  float* b_qin   = b_big  + SZ_BIG;      // later: attn out; later: conv out
  float* b_kvin  = b_qin  + SZ_Q;        // later: attn proj (+fused residual)
  float* b_kvout = b_kvin + SZ_KVIN;     // later: t1, t2
  float* b_qout  = b_kvout+ SZ_KVO;
  float* f32_end = b_qout + SZ_Q;

  float* b_attn  = b_qin;
  float* b_aproj = b_kvin;
  float* b_res   = b_aproj;              // result fused into aproj rows [0,NTOK)
  float* b_t1    = b_kvout;
  float* b_t2    = b_kvout + SZ_TOK;
  float* b_conv  = b_qin;
  float* b_res3  = b_lnx;

  // bf16 (ushort) region: weights + two activation scratches
  unsigned short* us = (unsigned short*)f32_end;
  unsigned short* wb_f0_in  = us;
  unsigned short* wb_f0_out = wb_f0_in  + (size_t)F_ * D_;
  unsigned short* wb_f1_in  = wb_f0_out + (size_t)F_ * D_;
  unsigned short* wb_f1_out = wb_f1_in  + (size_t)F_ * D_;
  unsigned short* wb_q      = wb_f1_out + (size_t)F_ * D_;
  unsigned short* wb_kv     = wb_q      + (size_t)D_ * D_;
  unsigned short* wb_out    = wb_kv     + (size_t)2 * D_ * D_;
  unsigned short* wb_pre    = wb_out    + (size_t)D_ * D_;
  unsigned short* wb_post   = wb_pre    + (size_t)2 * D_ * D_;
  unsigned short* abf_small = wb_post   + (size_t)D_ * D_;   // SZ_ABF elems (padded)
  unsigned short* abf_big   = abf_small + SZ_ABF;            // SZ_BIG elems
  // total ws ~ 353 MB

  const int rowsTok = NTOK * B_, rowsQ = QTOK * B_, rowsKV = KLTOK * B_;
  const unsigned EW = 256;
  const int BIGROWS = 1 << 30;   // "residual everywhere" bound

  // ---- weights -> bf16 (once per launch) ----
  tobf16_kernel<<<blks((size_t)F_ * D_), EW, 0, stream>>>(f0_w_in,  wb_f0_in,  (size_t)F_ * D_);
  tobf16_kernel<<<blks((size_t)F_ * D_), EW, 0, stream>>>(f0_w_out, wb_f0_out, (size_t)F_ * D_);
  tobf16_kernel<<<blks((size_t)F_ * D_), EW, 0, stream>>>(f1_w_in,  wb_f1_in,  (size_t)F_ * D_);
  tobf16_kernel<<<blks((size_t)F_ * D_), EW, 0, stream>>>(f1_w_out, wb_f1_out, (size_t)F_ * D_);
  tobf16_kernel<<<blks((size_t)D_ * D_), EW, 0, stream>>>(wq,   wb_q,   (size_t)D_ * D_);
  tobf16_kernel<<<blks((size_t)2 * D_ * D_), EW, 0, stream>>>(wkv, wb_kv, (size_t)2 * D_ * D_);
  tobf16_kernel<<<blks((size_t)D_ * D_), EW, 0, stream>>>(wout, wb_out, (size_t)D_ * D_);
  tobf16_kernel<<<blks((size_t)2 * D_ * D_), EW, 0, stream>>>(c_pre_w, wb_pre, (size_t)2 * D_ * D_);
  tobf16_kernel<<<blks((size_t)D_ * D_), EW, 0, stream>>>(c_post_w, wb_post, (size_t)D_ * D_);

  // ---- inp = concat(right_context, utterance) ----
  (void)hipMemcpyAsync(b_inp, rc, (size_t)R_ * B_ * D_ * 4, hipMemcpyDeviceToDevice, stream);
  (void)hipMemcpyAsync(b_inp + (size_t)R_ * B_ * D_, utter, (size_t)T_ * B_ * D_ * 4,
                       hipMemcpyDeviceToDevice, stream);

  // ---- ffn0 ----
  ln_kernel<<<rowsTok, EW, 0, stream>>>(b_inp, f0_ln_g, f0_ln_b, b_lnx, abf_small, rowsTok);
  gemm_bf16_wmma<<<gemm_grid(rowsTok, F_), EW, 0, stream>>>(
      abf_small, wb_f0_in, f0_b_in, nullptr, 0, nullptr, abf_big, rowsTok, F_, D_, 1.f, 1);
  gemm_bf16_wmma<<<gemm_grid(rowsTok, D_), EW, 0, stream>>>(
      abf_big, wb_f0_out, f0_b_out, b_inp, BIGROWS, b_ffn0, nullptr, rowsTok, D_, F_, 0.5f, 0);

  // ---- ln_in, q/kv inputs ----
  ln_kernel<<<rowsTok, EW, 0, stream>>>(b_ffn0, ln_in_g, ln_in_b, b_lnx, nullptr, rowsTok);
  (void)hipMemcpyAsync(b_qin, b_lnx, SZ_TOK * 4, hipMemcpyDeviceToDevice, stream);
  summary_kernel<<<blks(NS_ * B_ * D_), EW, 0, stream>>>(b_lnx, b_qin);
  (void)hipMemcpyAsync(b_kvin, mems, (size_t)M_ * B_ * D_ * 4, hipMemcpyDeviceToDevice, stream);
  (void)hipMemcpyAsync(b_kvin + (size_t)M_ * B_ * D_, b_lnx, SZ_TOK * 4,
                       hipMemcpyDeviceToDevice, stream);

  // ---- attention ----
  tobf16_kernel<<<blks(SZ_Q), EW, 0, stream>>>(b_qin, abf_small, SZ_Q);
  gemm_bf16_wmma<<<gemm_grid(rowsQ, D_), EW, 0, stream>>>(
      abf_small, wb_q, bq, nullptr, 0, b_qout, nullptr, rowsQ, D_, D_, 1.f, 0);
  tobf16_kernel<<<blks(SZ_KVIN), EW, 0, stream>>>(b_kvin, abf_small, SZ_KVIN);
  gemm_bf16_wmma<<<gemm_grid(rowsKV, 2 * D_), EW, 0, stream>>>(
      abf_small, wb_kv, bkv, nullptr, 0, b_kvout, nullptr, rowsKV, 2 * D_, D_, 1.f, 0);
  attention_kernel<<<dim3(QTOK, B_), EW, 0, stream>>>(
      b_qout, b_kvout, amask, lengths, b_attn, abf_small, QTOK, KLTOK);
  // out-projection with fused residual: rows [0,NTOK*B) get +ffn0 (result),
  // summary rows stay raw for the mems clip.
  gemm_bf16_wmma<<<gemm_grid(rowsQ, D_), EW, 0, stream>>>(
      abf_small, wb_out, bout, b_ffn0, rowsTok, b_aproj, nullptr, rowsQ, D_, D_, 1.f, 0);
  clip_mems_kernel<<<blks(S_ * B_ * D_), EW, 0, stream>>>(
      b_aproj, out + (size_t)(T_ + R_) * B_ * D_);

  // ---- conv module ----
  ln_kernel<<<rowsTok, EW, 0, stream>>>(b_res, c_pre_ln_g, c_pre_ln_b, b_t1, abf_small, rowsTok);
  gemm_bf16_wmma<<<gemm_grid(rowsTok, 2 * D_), EW, 0, stream>>>(
      abf_small, wb_pre, c_pre_b, nullptr, 0, b_big, nullptr, rowsTok, 2 * D_, D_, 1.f, 0);
  glu_kernel<<<blks(SZ_TOK), EW, 0, stream>>>(b_big, b_t2, rowsTok);
  conv_utt_kernel<<<blks((size_t)T_ * B_ * D_), EW, 0, stream>>>(b_t2, c_conv_w, c_conv_b, b_t1);
  conv_rc_kernel<<<blks((size_t)R_ * B_ * D_), EW, 0, stream>>>(b_t2, c_conv_w, c_conv_b, b_t1);
  ln_kernel<<<rowsTok, EW, 0, stream>>>(b_t1, c_post_ln_g, c_post_ln_b, b_t2, nullptr, rowsTok);
  silu_kernel<<<blks(SZ_TOK), EW, 0, stream>>>(b_t2, abf_small, SZ_TOK);
  gemm_bf16_wmma<<<gemm_grid(rowsTok, D_), EW, 0, stream>>>(
      abf_small, wb_post, c_post_b, b_res, BIGROWS, b_conv, nullptr, rowsTok, D_, D_, 1.f, 0);

  // ---- ffn1 ----
  ln_kernel<<<rowsTok, EW, 0, stream>>>(b_conv, f1_ln_g, f1_ln_b, b_t1, abf_small, rowsTok);
  gemm_bf16_wmma<<<gemm_grid(rowsTok, F_), EW, 0, stream>>>(
      abf_small, wb_f1_in, f1_b_in, nullptr, 0, nullptr, abf_big, rowsTok, F_, D_, 1.f, 1);
  gemm_bf16_wmma<<<gemm_grid(rowsTok, D_), EW, 0, stream>>>(
      abf_big, wb_f1_out, f1_b_out, b_conv, BIGROWS, b_res3, nullptr, rowsTok, D_, F_, 0.5f, 0);

  // ---- final LN + scatter outputs ----
  ln_kernel<<<rowsTok, EW, 0, stream>>>(b_res3, ln_out_g, ln_out_b, b_t2, nullptr, rowsTok);
  (void)hipMemcpyAsync(out, b_t2 + (size_t)R_ * B_ * D_, (size_t)T_ * B_ * D_ * 4,
                       hipMemcpyDeviceToDevice, stream);
  (void)hipMemcpyAsync(out + (size_t)T_ * B_ * D_, b_t2, (size_t)R_ * B_ * D_ * 4,
                       hipMemcpyDeviceToDevice, stream);
}
